// MultiLevelGraphLayer_4879082848577
// MI455X (gfx1250) — compile-verified, matrix operand order checked by reading, not run on hardware
//
#include <hip/hip_runtime.h>

#define D_  256
#define H_  8
#define DH_ 32
#define NH_ 1024
#define NL_ 65536
#define EH_ 16384
#define EL_ 262144

typedef __attribute__((ext_vector_type(16))) __bf16 v16bf;
typedef __attribute__((ext_vector_type(8)))  float  v8f;

union BF16Frag { v16bf v; unsigned int u[8]; };
union AccFrag  { v8f   v; float        f[8]; };

__device__ __forceinline__ unsigned short f2bf(float f) {
  union { float f; unsigned int u; } x; x.f = f;
  unsigned int r = x.u + 0x7FFFu + ((x.u >> 16) & 1u);
  return (unsigned short)(r >> 16);
}

__device__ __forceinline__ float waveSum(float v) {
  v += __shfl_xor(v, 1, 32);
  v += __shfl_xor(v, 2, 32);
  v += __shfl_xor(v, 4, 32);
  v += __shfl_xor(v, 8, 32);
  v += __shfl_xor(v, 16, 32);
  return v;
}

__device__ __forceinline__ void atomicMaxFloat(float* addr, float val) {
  if (val >= 0.f) atomicMax((int*)addr, __float_as_int(val));
  else            atomicMin((unsigned int*)addr, (unsigned int)__float_as_int(val));
}

// ---------------------------------------------------------------------------
// Generic bf16 WMMA GEMM: C[M,N] = A[M,K] @ B[K,N] (+bias) (+add), batched.
// Block tile 128x128, 8 waves (2 row groups x 4 col groups); each wave owns
// 64x32 = 8 wmma accumulators. A tile streamed HBM->LDS with async copies
// (double buffered); B tile register-staged and packed into (k,k+1) pairs.
// ---------------------------------------------------------------------------
struct GemmArgs {
  const unsigned short* A; const unsigned short* B;
  const float* bias; const float* add;
  float* Cf; unsigned short* Cb;
  int lda, ldb, ldc, M, N, K;
  long long sA, sB, sC;
};

#define BM 128
#define BN 128
#define BK 32

__global__ __launch_bounds__(256) void gemm_bf16(GemmArgs g) {
  __shared__ uint4 As4[2][BM * BK / 8];             // 2 x 8 KB (128 rows x 32 bf16)
  __shared__ unsigned int Bs[2][(BK / 2) * BN];     // 2 x 8 KB (16 kp x 128 packed)

  const int tid  = threadIdx.x;
  const int lane = tid & 31;
  const int wave = tid >> 5;
  const int wm   = wave >> 2;          // 0..1 : 64-row group
  const int wn   = wave & 3;           // 0..3 : 32-col group
  const int z    = blockIdx.z;
  const unsigned short* A = g.A + (long long)z * g.sA;
  const unsigned short* B = g.B + (long long)z * g.sB;
  const int row0 = blockIdx.x * BM;
  const int col0 = blockIdx.y * BN;

  // ---- async HBM -> LDS staging of the A tile (512 x 16B chunks) ----
  auto stageA = [&](int k0, int buf) {
#pragma unroll
    for (int i = 0; i < 2; ++i) {
      int c = tid + i * 256;
      int r = c >> 2, seg = c & 3;
      const unsigned short* src = A + (long long)(row0 + r) * g.lda + k0 + seg * 8;
      unsigned int ldsoff = (unsigned int)(unsigned long long)(&As4[buf][c]);
      asm volatile("global_load_async_to_lds_b128 %0, %1, off"
                   :: "v"(ldsoff), "v"((unsigned long long)src) : "memory");
    }
  };

  // ---- register-staged B tile: 16 kp rows x 16 col segs, one per thread ----
  uint4 be, bo;
  const int bkp  = tid >> 4;           // 0..15
  const int bseg = tid & 15;           // 8 cols each
  auto loadB = [&](int k0) {
    int c = col0 + bseg * 8;
    be = {0, 0, 0, 0}; bo = {0, 0, 0, 0};
    if (c < g.N) {
      be = *(const uint4*)(B + (long long)(k0 + bkp * 2)     * g.ldb + c);
      bo = *(const uint4*)(B + (long long)(k0 + bkp * 2 + 1) * g.ldb + c);
    }
  };
  auto storeB = [&](int buf) {
    const unsigned short* eh = (const unsigned short*)&be;
    const unsigned short* oh = (const unsigned short*)&bo;
    unsigned int* dst = &Bs[buf][bkp * BN + bseg * 8];
#pragma unroll
    for (int j = 0; j < 8; ++j)
      dst[j] = (unsigned int)eh[j] | ((unsigned int)oh[j] << 16);
  };

  AccFrag acc[4][2];
#pragma unroll
  for (int m = 0; m < 4; ++m)
#pragma unroll
    for (int t = 0; t < 2; ++t)
#pragma unroll
      for (int i = 0; i < 8; ++i) acc[m][t].f[i] = 0.f;

  const int nk = g.K / BK;
  stageA(0, 0);
  loadB(0);
  storeB(0);
  asm volatile("s_wait_asynccnt 0x0" ::: "memory");
  __syncthreads();

  for (int ks = 0; ks < nk; ++ks) {
    const int buf = ks & 1;
    if (ks + 1 < nk) {
      stageA((ks + 1) * BK, buf ^ 1);   // in flight during WMMA below
      loadB((ks + 1) * BK);             // in flight during WMMA below
    }
    const unsigned int* Asu = (const unsigned int*)As4[buf];
    const unsigned int* Bsu = Bs[buf];
    const int half = lane >> 4;

    BF16Frag bf[2];
#pragma unroll
    for (int t = 0; t < 2; ++t) {
      int c = wn * 32 + (t << 4) + (lane & 15);
#pragma unroll
      for (int v = 0; v < 8; ++v) {
        int p = v & 3, gg = v >> 2;
        bf[t].u[v] = Bsu[(gg * 8 + half * 4 + p) * BN + c];
      }
    }
#pragma unroll
    for (int m = 0; m < 4; ++m) {
      BF16Frag af;
      int r = wm * 64 + (m << 4) + (lane & 15);
#pragma unroll
      for (int v = 0; v < 8; ++v) {
        int p = v & 3, gg = v >> 2;
        af.u[v] = Asu[r * 16 + gg * 8 + half * 4 + p];
      }
      acc[m][0].v = __builtin_amdgcn_wmma_f32_16x16x32_bf16(
          false, af.v, false, bf[0].v, (short)0, acc[m][0].v, false, false);
      acc[m][1].v = __builtin_amdgcn_wmma_f32_16x16x32_bf16(
          false, af.v, false, bf[1].v, (short)0, acc[m][1].v, false, false);
    }
    if (ks + 1 < nk) storeB(buf ^ 1);
    asm volatile("s_wait_asynccnt 0x0" ::: "memory");
    __syncthreads();
  }

  // ---- epilogue ----
  const long long zc = (long long)z * g.sC;
  const int half = lane >> 4;
  const int n_in = lane & 15;
#pragma unroll
  for (int m = 0; m < 4; ++m) {
    int grb = row0 + wm * 64 + (m << 4) + 8 * half;
#pragma unroll
    for (int t = 0; t < 2; ++t) {
      int gc = col0 + wn * 32 + (t << 4) + n_in;
      if (gc >= g.N) continue;
      float bia = g.bias ? g.bias[gc] : 0.f;
#pragma unroll
      for (int v = 0; v < 8; ++v) {
        long long ci = (long long)(grb + v) * g.ldc + gc;
        float val = acc[m][t].f[v] + bia;
        if (g.add) val += g.add[ci];
        if (g.Cf) g.Cf[zc + ci] = val;
        if (g.Cb) g.Cb[zc + ci] = f2bf(val);
      }
    }
  }
}

// ---------------------------------------------------------------------------
// Elementwise / reduction kernels
// ---------------------------------------------------------------------------
__global__ void fill_f32(float* p, float v, long long n) {
  long long i = (long long)blockIdx.x * 256 + threadIdx.x;
  if (i < n) p[i] = v;
}

__global__ void cvt_f32_bf16(const float* src, unsigned short* dst, int n) {
  int i = blockIdx.x * 256 + threadIdx.x;
  if (i < n) dst[i] = f2bf(src[i]);
}

__global__ __launch_bounds__(256) void ln_kernel(const float* x, const float* g,
                                                 const float* b, float* outf,
                                                 unsigned short* outb, int nrows) {
  int row  = blockIdx.x * 8 + (threadIdx.x >> 5);
  int lane = threadIdx.x & 31;
  if (row >= nrows) return;
  const float* src = x + (long long)row * D_ + lane * 8;
  float v[8];
#pragma unroll
  for (int j = 0; j < 8; ++j) v[j] = src[j];
  float s = 0.f;
#pragma unroll
  for (int j = 0; j < 8; ++j) s += v[j];
  s = waveSum(s);
  float mean = s * (1.f / D_);
  float q = 0.f;
#pragma unroll
  for (int j = 0; j < 8; ++j) { float d = v[j] - mean; q += d * d; }
  q = waveSum(q);
  float rstd = rsqrtf(q * (1.f / D_) + 1e-5f);
  const float* gg = g + lane * 8;
  const float* bb = b + lane * 8;
  long long o = (long long)row * D_ + lane * 8;
#pragma unroll
  for (int j = 0; j < 8; ++j) {
    float y = (v[j] - mean) * rstd * gg[j] + bb[j];
    if (outf) outf[o + j] = y;
    outb[o + j] = f2bf(y);
  }
}

__global__ void gin_agg(const float* he, const int* src, const int* dst, float* agg) {
  int e = blockIdx.x, t = threadIdx.x;
  int s = src[e], d = dst[e];
  atomicAdd(&agg[(long long)d * D_ + t], he[(long long)s * D_ + t]);
}

__global__ void gin_pre(const float* he, const float* agg, const float* epsp,
                        unsigned short* out, int n) {
  int i = blockIdx.x * 256 + threadIdx.x;
  if (i < n) out[i] = f2bf((1.f + *epsp) * he[i] + agg[i]);
}

__global__ void transpose_heads(const unsigned short* kb, unsigned short* kt) {
  int idx = blockIdx.x * 256 + threadIdx.x;   // H*DH*NH = 262144
  int n = idx & (NH_ - 1);
  int hd = idx >> 10;                         // h*32+d
  kt[(long long)hd * NH_ + n] = kb[(long long)n * D_ + hd];
}

__global__ __launch_bounds__(256) void softmax_rows(float* logits, unsigned short* out,
                                                    float scale) {
  __shared__ float red[256];
  long long row = blockIdx.x;
  float* src = logits + row * NH_;
  int tid = threadIdx.x;
  float vals[4];
  float mx = -1e30f;
#pragma unroll
  for (int i = 0; i < 4; ++i) {
    vals[i] = src[tid + i * 256] * scale;
    mx = fmaxf(mx, vals[i]);
  }
  red[tid] = mx; __syncthreads();
  for (int s = 128; s > 0; s >>= 1) {
    if (tid < s) red[tid] = fmaxf(red[tid], red[tid + s]);
    __syncthreads();
  }
  mx = red[0]; __syncthreads();
  float sum = 0.f;
#pragma unroll
  for (int i = 0; i < 4; ++i) { vals[i] = __expf(vals[i] - mx); sum += vals[i]; }
  red[tid] = sum; __syncthreads();
  for (int s = 128; s > 0; s >>= 1) {
    if (tid < s) red[tid] += red[tid + s];
    __syncthreads();
  }
  float inv = 1.f / red[0];
  unsigned short* dst = out + row * 2048;     // in-place over f32 row (pitch 2048 u16)
#pragma unroll
  for (int i = 0; i < 4; ++i) dst[tid + i * 256] = f2bf(vals[i] * inv);
}

__global__ __launch_bounds__(256) void edge_logits(const float* tq, const float* tk,
                                                   const int* ls, const int* lt,
                                                   float* lg, float* mx, int ne) {
  int e = blockIdx.x * 8 + (threadIdx.x >> 5);
  if (e >= ne) return;
  int lane = threadIdx.x & 31;
  int s = ls[e], t = lt[e];
  const float* q = tq + (long long)t * D_ + lane * 8;
  const float* k = tk + (long long)s * D_ + lane * 8;
  float dot = 0.f;
#pragma unroll
  for (int j = 0; j < 8; ++j) dot += q[j] * k[j];
  dot += __shfl_xor(dot, 1, 32);
  dot += __shfl_xor(dot, 2, 32);                 // 4 lanes = one head (32 dims)
  if ((lane & 3) == 0) {
    int h = lane >> 2;
    float v = dot * 0.17677669529663687f;        // 1/sqrt(32)
    lg[(long long)e * H_ + h] = v;
    atomicMaxFloat(&mx[(long long)t * H_ + h], v);
  }
}

__global__ void edge_exp(const float* lg, const float* mx, const int* lt,
                         float* p, float* den, long long n) {
  long long i = (long long)blockIdx.x * 256 + threadIdx.x;
  if (i >= n) return;
  long long e = i >> 3;
  int h = (int)(i & 7);
  int t = lt[e];
  float v = __expf(lg[i] - mx[(long long)t * H_ + h]);
  p[i] = v;
  atomicAdd(&den[(long long)t * H_ + h], v);
}

__global__ void edge_scatter(const float* p, const float* den, const float* tv,
                             const int* ls, const int* lt, float* msg) {
  int e = blockIdx.x, t = threadIdx.x;
  int s = ls[e], d = lt[e];
  int h = t >> 5;
  float alpha = p[(long long)e * H_ + h] / den[(long long)d * H_ + h];
  atomicAdd(&msg[(long long)d * D_ + t], alpha * tv[(long long)s * D_ + t]);
}

__global__ void xg_scatter(const float* low, const int* batch, float* xg, float* cnt) {
  int i = blockIdx.x, t = threadIdx.x;
  int b = batch[i];
  atomicAdd(&xg[(long long)b * D_ + t], low[(long long)i * D_ + t]);
  if (t == 0) atomicAdd(&cnt[b], 1.0f);
}

__global__ void xg_norm(float* xg, const float* cnt, unsigned short* xgb, int n) {
  int i = blockIdx.x * 256 + threadIdx.x;
  if (i >= n) return;
  float c = fmaxf(cnt[i >> 8], 1.0f);
  float v = xg[i] / c;
  xg[i] = v;
  xgb[i] = f2bf(v);
}

__global__ __launch_bounds__(256) void finish_rows(const float* qv, const float* kv,
                                                   const float* vv, const int* gidx,
                                                   float* out, int nrows, float sd) {
  int row  = blockIdx.x * 8 + (threadIdx.x >> 5);
  int lane = threadIdx.x & 31;
  if (row >= nrows) return;
  int kr = gidx ? gidx[row] : row;
  const float* q = qv + (long long)row * D_ + lane * 8;
  const float* k = kv + (long long)kr * D_ + lane * 8;
  float dot = 0.f;
#pragma unroll
  for (int j = 0; j < 8; ++j) dot += q[j] * k[j];
  float w = waveSum(dot) * sd;
  const float* v = vv + (long long)row * D_ + lane * 8;
  float* o = out + (long long)row * D_ + lane * 8;
#pragma unroll
  for (int j = 0; j < 8; ++j) {
    float x = w * v[j];
    o[j] = 0.5f * x * (1.0f + erff(x * 0.70710678118654752f));
  }
}

// ---------------------------------------------------------------------------
// Host side
// ---------------------------------------------------------------------------
static inline void gemm(hipStream_t st, const unsigned short* A, int lda, long long sA,
                        const unsigned short* B, int ldb, long long sB,
                        const float* bias, const float* add,
                        float* Cf, unsigned short* Cb, int ldc, long long sC,
                        int M, int N, int K, int batch) {
  GemmArgs g;
  g.A = A; g.B = B; g.bias = bias; g.add = add; g.Cf = Cf; g.Cb = Cb;
  g.lda = lda; g.ldb = ldb; g.ldc = ldc; g.M = M; g.N = N; g.K = K;
  g.sA = sA; g.sB = sB; g.sC = sC;
  dim3 grid(M / BM, (N + BN - 1) / BN, batch);
  gemm_bf16<<<grid, 256, 0, st>>>(g);
}

extern "C" void kernel_launch(void* const* d_in, const int* in_sizes, int n_in,
                              void* d_out, int out_size, void* d_ws, size_t ws_size,
                              hipStream_t stream) {
  (void)in_sizes; (void)n_in; (void)out_size; (void)ws_size;
  const float* high_in = (const float*)d_in[0];
  const float* low_in  = (const float*)d_in[1];
  const int*   hEdge   = (const int*)d_in[2];     // [2, EH]: src then dst
  const int*   lEdge   = (const int*)d_in[3];     // [2, EL]
  const int*   lBatch  = (const int*)d_in[4];
  const float* ln_g    = (const float*)d_in[5];
  const float* ln_b    = (const float*)d_in[6];
  const float* ginEps  = (const float*)d_in[7];
  const float* wsrc[15] = {
    (const float*)d_in[8],                         // gin_w
    (const float*)d_in[10], (const float*)d_in[12],
    (const float*)d_in[14], (const float*)d_in[16],// mha wq wk wv wo
    (const float*)d_in[18], (const float*)d_in[20],
    (const float*)d_in[22], (const float*)d_in[24],// tc wq wk wv wskip
    (const float*)d_in[26], (const float*)d_in[27], (const float*)d_in[28],
    (const float*)d_in[29], (const float*)d_in[30], (const float*)d_in[31]
  };
  const float* gin_b  = (const float*)d_in[9];
  const float* mha_bq = (const float*)d_in[11];
  const float* mha_bk = (const float*)d_in[13];
  const float* mha_bv = (const float*)d_in[15];
  const float* mha_bo = (const float*)d_in[17];
  const float* tc_bq  = (const float*)d_in[19];
  const float* tc_bk  = (const float*)d_in[21];
  const float* tc_bv  = (const float*)d_in[23];
  const float* tc_bsk = (const float*)d_in[25];

  // ---- workspace carve-up ----
  char* base = (char*)d_ws;
  size_t off = 0;
  auto alloc = [&](size_t bytes) -> void* {
    void* r = base + off;
    off += (bytes + 255) & ~(size_t)255;
    return r;
  };
  const size_t NHD = (size_t)NH_ * D_, NLD = (size_t)NL_ * D_;
  unsigned short* wbf[15];
  for (int i = 0; i < 15; ++i) wbf[i] = (unsigned short*)alloc(D_ * D_ * 2);
  float*          he_f   = (float*)alloc(NHD * 4);
  unsigned short* he_b   = (unsigned short*)alloc(NHD * 2);
  unsigned short* le_b   = (unsigned short*)alloc(NLD * 2);
  float*          agg    = (float*)alloc(NHD * 4);
  unsigned short* ginin  = (unsigned short*)alloc(NHD * 2);
  float*          gin_f  = (float*)alloc(NHD * 4);
  unsigned short* Qb     = (unsigned short*)alloc(NHD * 2);
  unsigned short* Kb     = (unsigned short*)alloc(NHD * 2);
  unsigned short* Vb     = (unsigned short*)alloc(NHD * 2);
  unsigned short* Ktb    = (unsigned short*)alloc(NHD * 2);
  float*          logits = (float*)alloc((size_t)H_ * NH_ * NH_ * 4);  // 32 MB
  unsigned short* mhh_b  = (unsigned short*)alloc(NHD * 2);
  float*          high_f = (float*)alloc(NHD * 4);
  unsigned short* high_b = (unsigned short*)alloc(NHD * 2);
  float*          tq_f   = (float*)alloc(NLD * 4);   // later: msg, then lowq
  float*          tk_f   = (float*)alloc(NLD * 4);   // later: lowv
  float*          tv_f   = (float*)alloc(NLD * 4);   // later: low_f
  float*          mx     = (float*)alloc((size_t)NL_ * H_ * 4);
  float*          den    = (float*)alloc((size_t)NL_ * H_ * 4);
  float*          xg     = (float*)alloc(NHD * 4);
  unsigned short* xg_b   = (unsigned short*)alloc(NHD * 2);
  float*          cnt    = (float*)alloc(NH_ * 4);
  float*          highq  = (float*)alloc(NHD * 4);
  float*          xgk    = (float*)alloc(NHD * 4);
  float*          highv  = (float*)alloc(NHD * 4);
  float*          hk     = (float*)alloc(NHD * 4);
  // time-sliced aliases
  float*          lg    = logits;                    // edge logits (8 MB)
  float*          pbuf  = logits + (size_t)EL_ * H_; // exp values (8 MB)
  unsigned short* low_b = (unsigned short*)logits;   // after edge passes (32 MB)
  float*          msg   = tq_f;                      // after edge pass a
  float*          low_f = tv_f;                      // after edge pass c
  float*          lowq  = tq_f;                      // after low GEMM
  float*          lowv  = tk_f;
  float*          out_high = (float*)d_out;
  float*          out_low  = (float*)d_out + NHD;

  const unsigned short* attn_b = (const unsigned short*)logits; // in-place bf16

  // ---- 1) weights to bf16 ----
  for (int i = 0; i < 15; ++i)
    cvt_f32_bf16<<<(D_ * D_ + 255) / 256, 256, 0, stream>>>(wsrc[i], wbf[i], D_ * D_);

  // ---- 2) layernorms ----
  ln_kernel<<<NH_ / 8, 256, 0, stream>>>(high_in, ln_g, ln_b, he_f, he_b, NH_);
  ln_kernel<<<NL_ / 8, 256, 0, stream>>>(low_in,  ln_g, ln_b, nullptr, le_b, NL_);

  // ---- 3) GIN ----
  fill_f32<<<(int)((NHD + 255) / 256), 256, 0, stream>>>(agg, 0.f, (long long)NHD);
  gin_agg<<<EH_, 256, 0, stream>>>(he_f, hEdge, hEdge + EH_, agg);
  gin_pre<<<(int)((NHD + 255) / 256), 256, 0, stream>>>(he_f, agg, ginEps, ginin, (int)NHD);
  gemm(stream, ginin, D_, 0, wbf[0], D_, 0, gin_b, nullptr,
       gin_f, nullptr, D_, 0, NH_, D_, D_, 1);

  // ---- 4) MHA ----
  gemm(stream, he_b, D_, 0, wbf[1], D_, 0, mha_bq, nullptr, nullptr, Qb, D_, 0, NH_, D_, D_, 1);
  gemm(stream, he_b, D_, 0, wbf[2], D_, 0, mha_bk, nullptr, nullptr, Kb, D_, 0, NH_, D_, D_, 1);
  gemm(stream, he_b, D_, 0, wbf[3], D_, 0, mha_bv, nullptr, nullptr, Vb, D_, 0, NH_, D_, D_, 1);
  transpose_heads<<<(int)(NHD / 256), 256, 0, stream>>>(Kb, Ktb);
  // logits[h] = Q_h (1024x32) @ K_h^T (32x1024)
  gemm(stream, Qb, D_, DH_, Ktb, NH_, (long long)DH_ * NH_, nullptr, nullptr,
       logits, nullptr, NH_, (long long)NH_ * NH_, NH_, NH_, DH_, H_);
  softmax_rows<<<H_ * NH_, 256, 0, stream>>>(logits, (unsigned short*)logits,
                                             0.17677669529663687f);
  // mh_heads[h] = P_h (1024x1024) @ V_h (1024x32)
  gemm(stream, attn_b, 2 * NH_, (long long)NH_ * 2 * NH_, Vb, D_, DH_, nullptr, nullptr,
       nullptr, mhh_b, D_, DH_, NH_, DH_, NH_, H_);
  // high = mh_heads @ Wo + bo + gin
  gemm(stream, mhh_b, D_, 0, wbf[4], D_, 0, mha_bo, gin_f,
       high_f, high_b, D_, 0, NH_, D_, D_, 1);

  // ---- 5) TransformerConv on low graph ----
  gemm(stream, le_b, D_, 0, wbf[5], D_, 0, tc_bq, nullptr, tq_f, nullptr, D_, 0, NL_, D_, D_, 1);
  gemm(stream, le_b, D_, 0, wbf[6], D_, 0, tc_bk, nullptr, tk_f, nullptr, D_, 0, NL_, D_, D_, 1);
  gemm(stream, le_b, D_, 0, wbf[7], D_, 0, tc_bv, nullptr, tv_f, nullptr, D_, 0, NL_, D_, D_, 1);
  fill_f32<<<(NL_ * H_ + 255) / 256, 256, 0, stream>>>(mx, -1e30f, (long long)NL_ * H_);
  edge_logits<<<EL_ / 8, 256, 0, stream>>>(tq_f, tk_f, lEdge, lEdge + EL_, lg, mx, EL_);
  fill_f32<<<(NL_ * H_ + 255) / 256, 256, 0, stream>>>(den, 0.f, (long long)NL_ * H_);
  edge_exp<<<(int)(((long long)EL_ * H_ + 255) / 256), 256, 0, stream>>>(
      lg, mx, lEdge + EL_, pbuf, den, (long long)EL_ * H_);
  fill_f32<<<(int)((NLD + 255) / 256), 256, 0, stream>>>(msg, 0.f, (long long)NLD);
  edge_scatter<<<EL_, 256, 0, stream>>>(pbuf, den, tv_f, lEdge, lEdge + EL_, msg);
  // low = msg + le @ Wskip + bskip
  gemm(stream, le_b, D_, 0, wbf[8], D_, 0, tc_bsk, msg,
       low_f, low_b, D_, 0, NL_, D_, D_, 1);

  // ---- 6) cross message passing ----
  fill_f32<<<(int)((NHD + 255) / 256), 256, 0, stream>>>(xg, 0.f, (long long)NHD);
  fill_f32<<<(NH_ + 255) / 256, 256, 0, stream>>>(cnt, 0.f, (long long)NH_);
  xg_scatter<<<NL_, 256, 0, stream>>>(low_f, lBatch, xg, cnt);
  xg_norm<<<(int)((NHD + 255) / 256), 256, 0, stream>>>(xg, cnt, xg_b, (int)NHD);
  gemm(stream, high_b, D_, 0, wbf[9],  D_, 0, nullptr, nullptr, highq, nullptr, D_, 0, NH_, D_, D_, 1);
  gemm(stream, xg_b,   D_, 0, wbf[10], D_, 0, nullptr, nullptr, xgk,   nullptr, D_, 0, NH_, D_, D_, 1);
  gemm(stream, high_b, D_, 0, wbf[11], D_, 0, nullptr, nullptr, highv, nullptr, D_, 0, NH_, D_, D_, 1);
  gemm(stream, high_b, D_, 0, wbf[13], D_, 0, nullptr, nullptr, hk,    nullptr, D_, 0, NH_, D_, D_, 1);
  gemm(stream, low_b,  D_, 0, wbf[12], D_, 0, nullptr, nullptr, lowq,  nullptr, D_, 0, NL_, D_, D_, 1);
  gemm(stream, low_b,  D_, 0, wbf[14], D_, 0, nullptr, nullptr, lowv,  nullptr, D_, 0, NL_, D_, D_, 1);

  const float sd = 0.0625f; // 1/sqrt(256)
  finish_rows<<<NH_ / 8, 256, 0, stream>>>(highq, xgk, highv, nullptr, out_high, NH_, sd);
  finish_rows<<<NL_ / 8, 256, 0, stream>>>(lowq, hk, lowv, lBatch, out_low, NL_, sd);
}
